// TopKSelectorOTT_60232621359135
// MI455X (gfx1250) — matrix operand order repaired and checked by reading, not compiled, not run on hardware
//
#include <hip/hip_runtime.h>
#include <hip/hip_bf16.h>
#include <math.h>

typedef __attribute__((ext_vector_type(16))) __bf16 v16bf;
typedef __attribute__((ext_vector_type(8)))  __bf16 v8bf;
typedef __attribute__((ext_vector_type(8)))  float  v8f;
typedef __attribute__((ext_vector_type(4)))  int    v4i;

#define BATCH 8192
#define D_IN  512
#define HID   512
#define NSUB  64

#define INV_EPS   10.0f                 // 1/temperature
#define LOG_MARG  (-4.158883083359672f) // -ln(64)
#define N_ITERS   50

// ---------------- CDNA5 async global->LDS copy (16B), with fallback --------
#if __has_builtin(__builtin_amdgcn_global_load_async_to_lds_b128)
#define HAVE_ASYNC_LDS 1
#else
#define HAVE_ASYNC_LDS 0
#endif

typedef __attribute__((address_space(1))) v4i* gv4i_p;  // global int4*
typedef __attribute__((address_space(3))) v4i* lv4i_p;  // LDS int4*

static __device__ __forceinline__ void async_copy16(const __bf16* g,
                                                    __bf16* l) {
#if HAVE_ASYNC_LDS
  __builtin_amdgcn_global_load_async_to_lds_b128((gv4i_p)g, (lv4i_p)l, 0, 0);
#else
  *(v8bf*)l = *(const v8bf*)g;   // load + ds_store fallback
#endif
}

static __device__ __forceinline__ void wait_async0() {
#if HAVE_ASYNC_LDS
#if __has_builtin(__builtin_amdgcn_s_wait_asynccnt)
  __builtin_amdgcn_s_wait_asynccnt(0);
#else
  asm volatile("s_wait_asynccnt 0x0" ::: "memory");
#endif
#endif
}

static __device__ __forceinline__ void sched_fence() {
#if __has_builtin(__builtin_amdgcn_sched_barrier)
  __builtin_amdgcn_sched_barrier(0);   // nothing may be scheduled across
#endif
}

// ---------------- elementwise fp32 -> bf16 ----------------
__global__ void cvt_bf16_kernel(const float* __restrict__ in,
                                __bf16* __restrict__ out, int n) {
  int i = blockIdx.x * blockDim.x + threadIdx.x;
  if (i < n) out[i] = (__bf16)in[i];
}

// ---------------- transpose fp32 [K,N] -> bf16 [N,K] ----------------
__global__ void transpose_bf16_kernel(const float* __restrict__ in,
                                      __bf16* __restrict__ out, int K, int N) {
  int i = blockIdx.x * blockDim.x + threadIdx.x;
  if (i < K * N) {
    int n = i / K;
    int k = i - n * K;
    out[i] = (__bf16)in[(size_t)k * N + n];   // out[n*K+k] = in[k][n]
  }
}

// ---------------- bf16 WMMA GEMM: out = act(A @ B + bias) ----------------
// A: bf16 [M,K] row-major.  BT: bf16 [N,K] row-major (B transposed).
// Block = 256 threads = 8 waves stacked along M: block tile 128(M) x 64(N).
// All 8 waves share the same 64xBK slab of BT, staged in double-buffered LDS
// via async global->LDS copies (ASYNCcnt) while WMMAs consume the other
// buffer. Each wave: 16x64 tile = 4 f32 accumulators, K stepped by 64
// (2 WMMA k-slabs of 32 per step). A sched_barrier between the LDS fragment
// loads and the WMMAs keeps all 8 B fragments live so the matrix ops issue
// back-to-back behind a single DScnt wait.
template <bool TANH, typename OutT>
__global__ __launch_bounds__(256) void gemm_wmma_kernel(
    const __bf16* __restrict__ A, const __bf16* __restrict__ BT,
    const float* __restrict__ bias, OutT* __restrict__ out,
    int M, int N, int K) {
  constexpr int BK  = 64;
  constexpr int BKP = 72;   // padded row stride (bf16): conflict-free ds reads
  __shared__ __bf16 bsh[2][64 * BKP];

  const int lane  = threadIdx.x & 31;
  const int wave  = threadIdx.x >> 5;
  const int tilesN = N >> 6;
  const int bm = blockIdx.x / tilesN;
  const int bn = blockIdx.x - bm * tilesN;
  const int rowtile = bm * 128 + wave * 16;   // this wave's 16 M-rows
  const int nbase   = bn * 64;

  const int r16   = lane & 15;
  const int khalf = lane >> 4;                // 0 or 1

  // A fragment: lane<16 -> K {0..7,16..23}; lane>=16 -> K {8..15,24..31}
  const __bf16* arow = A + (size_t)(rowtile + r16) * K;
  const int koffA = khalf * 8;

  // B staging: 64 rows x BK bf16 = 512 16B-chunks; 2 chunks per thread.
  const int c0 = threadIdx.x;
  const int rowb0 = c0 >> 3,          cob0 = (c0 & 7) * 8;
  const int rowb1 = (c0 + 256) >> 3,  cob1 = ((c0 + 256) & 7) * 8;
  const __bf16* bsrc0 = BT + (size_t)(nbase + rowb0) * K + cob0;
  const __bf16* bsrc1 = BT + (size_t)(nbase + rowb1) * K + cob1;

  // Stage k=0 into buffer 0.
  async_copy16(bsrc0, &bsh[0][rowb0 * BKP + cob0]);
  async_copy16(bsrc1, &bsh[0][rowb1 * BKP + cob1]);

  // A fragments for k=0.
  v16bf a0, a1;
  ((v8bf*)&a0)[0] = *(const v8bf*)(arow + koffA);
  ((v8bf*)&a0)[1] = *(const v8bf*)(arow + koffA + 16);
  ((v8bf*)&a1)[0] = *(const v8bf*)(arow + 32 + koffA);
  ((v8bf*)&a1)[1] = *(const v8bf*)(arow + 32 + koffA + 16);

  wait_async0();
  __syncthreads();

  v8f acc[4] = {v8f{}, v8f{}, v8f{}, v8f{}};
  int cur = 0;

  for (int k = 0; k < K; k += BK) {
    const int knext = k + BK;
    v16bf a0n, a1n;
    if (knext < K) {
      // Prefetch next A fragments into fresh registers.
      ((v8bf*)&a0n)[0] = *(const v8bf*)(arow + knext + koffA);
      ((v8bf*)&a0n)[1] = *(const v8bf*)(arow + knext + koffA + 16);
      ((v8bf*)&a1n)[0] = *(const v8bf*)(arow + knext + 32 + koffA);
      ((v8bf*)&a1n)[1] = *(const v8bf*)(arow + knext + 32 + koffA + 16);
      // Async-stage next B slab into the other buffer.
      async_copy16(bsrc0 + knext, &bsh[cur ^ 1][rowb0 * BKP + cob0]);
      async_copy16(bsrc1 + knext, &bsh[cur ^ 1][rowb1 * BKP + cob1]);
    }

    // Load ALL 8 B fragments from LDS into distinct registers first.
    // B fragment: lane = column n (r16), K {0..7,16..23} (lane<16) or
    // {8..15,24..31} (lane>=16), matching A.
    v16bf bfr[8];
#pragma unroll
    for (int tb = 0; tb < 4; ++tb) {
      const __bf16* bp = &bsh[cur][(tb * 16 + r16) * BKP + khalf * 8];
      ((v8bf*)&bfr[2 * tb + 0])[0] = *(const v8bf*)(bp);
      ((v8bf*)&bfr[2 * tb + 0])[1] = *(const v8bf*)(bp + 16);
      ((v8bf*)&bfr[2 * tb + 1])[0] = *(const v8bf*)(bp + 32);
      ((v8bf*)&bfr[2 * tb + 1])[1] = *(const v8bf*)(bp + 48);
    }
    // Forbid the scheduler from sinking these loads into the WMMA chain:
    // one DScnt wait, then 8 back-to-back WMMAs.
    sched_fence();
#pragma unroll
    for (int tb = 0; tb < 4; ++tb) {
      acc[tb] = __builtin_amdgcn_wmma_f32_16x16x32_bf16(
          false, a0, false, bfr[2 * tb + 0], (short)0, acc[tb], false, false);
      acc[tb] = __builtin_amdgcn_wmma_f32_16x16x32_bf16(
          false, a1, false, bfr[2 * tb + 1], (short)0, acc[tb], false, false);
    }

    if (knext < K) {
      wait_async0();
      a0 = a0n;
      a1 = a1n;
    }
    __syncthreads();
    cur ^= 1;
  }

  // Epilogue. C/D layout: VGPR r, lane l -> M = r + 8*(l>=16), N = l&15.
  const int mbase = rowtile + khalf * 8;
#pragma unroll
  for (int tb = 0; tb < 4; ++tb) {
    const int ncol = nbase + tb * 16 + r16;
    const float bv = bias[ncol];
#pragma unroll
    for (int rr = 0; rr < 8; ++rr) {
      float v = acc[tb][rr] + bv;
      if (TANH) v = tanhf(v);
      out[(size_t)(mbase + rr) * N + ncol] = (OutT)v;
    }
  }
}

// ---------------- per-row Sinkhorn soft-rank + top-K mask ----------------
// One 64-thread block (2 waves) per batch row. Thread t owns row i=t (for the
// f-update) and column j=t (for the g-update). neg_cost cached in LDS with
// stride 65 so both row sweeps (bank = (t+j)&63) and column sweeps
// (bank = (i+t)&63) are conflict-free. F,G kept in the scaled (f/eps) domain
// so inner loops are pure add+exp (hardware v_exp_f32).
__global__ __launch_bounds__(64) void sinkhorn_mask_kernel(
    const float* __restrict__ scores, float* __restrict__ out) {
  __shared__ float zsh[NSUB];
  __shared__ float Fsh[NSUB];
  __shared__ float Gsh[NSUB];
  __shared__ float nc[NSUB * 65];

  const int t   = threadIdx.x;
  const int row = blockIdx.x;
  const float s = scores[(size_t)row * NSUB + t];

  zsh[t] = s;
  __syncthreads();
  float sum = 0.f, sumsq = 0.f;
#pragma unroll 8
  for (int i = 0; i < NSUB; ++i) {
    float v = zsh[i];
    sum += v;
    sumsq += v * v;
  }
  const float mean = sum * (1.0f / NSUB);
  float var = sumsq * (1.0f / NSUB) - mean * mean;
  var = fmaxf(var, 0.f);
  const float sd = sqrtf(var);
  // z = sigmoid((s - mean) / (sd + 1e-10))
  const float zt = 1.f / (1.f + __expf(-((s - mean) / (sd + 1e-10f))));
  __syncthreads();
  zsh[t] = zt;
  Fsh[t] = 0.f;
  Gsh[t] = 0.f;
  __syncthreads();

  // neg_cost[t][j] = -(z_t - j/63)^2 / eps
#pragma unroll 8
  for (int j = 0; j < NSUB; ++j) {
    float d = zt - (float)j * (1.0f / 63.0f);
    nc[t * 65 + j] = -d * d * INV_EPS;
  }
  __syncthreads();

  float Ft = 0.f;
  for (int it = 0; it < N_ITERS; ++it) {
    // G_t = logmarg - LSE_i(F_i + nc[i][t])
    float m = -1e30f;
#pragma unroll 8
    for (int i = 0; i < NSUB; ++i) m = fmaxf(m, Fsh[i] + nc[i * 65 + t]);
    float ssum = 0.f;
#pragma unroll 8
    for (int i = 0; i < NSUB; ++i) ssum += __expf(Fsh[i] + nc[i * 65 + t] - m);
    const float Gt = LOG_MARG - (m + __logf(ssum));
    Gsh[t] = Gt;
    __syncthreads();

    // F_t = logmarg - LSE_j(G_j + nc[t][j])
    m = -1e30f;
#pragma unroll 8
    for (int j = 0; j < NSUB; ++j) m = fmaxf(m, Gsh[j] + nc[t * 65 + j]);
    ssum = 0.f;
#pragma unroll 8
    for (int j = 0; j < NSUB; ++j) ssum += __expf(Gsh[j] + nc[t * 65 + j] - m);
    Ft = LOG_MARG - (m + __logf(ssum));
    __syncthreads();
    Fsh[t] = Ft;
    __syncthreads();
  }

  // rank_t = n * sum_j exp(F_t + G_j + nc[t][j]) * j
  float r = 0.f;
#pragma unroll 8
  for (int j = 0; j < NSUB; ++j)
    r += __expf(Ft + Gsh[j] + nc[t * 65 + j]) * (float)j;
  const float rank = (float)NSUB * r;
  const float inverted = (float)(NSUB - 1) - rank;
  const float mask =
      1.f / (1.f + __expf(-(inverted - (float)(NSUB - 8)) * 10.0f));
  out[(size_t)row * NSUB + t] = mask;
}

// ---------------- launch ----------------
extern "C" void kernel_launch(void* const* d_in, const int* in_sizes, int n_in,
                              void* d_out, int out_size, void* d_ws,
                              size_t ws_size, hipStream_t stream) {
  const float* x    = (const float*)d_in[0];
  const float* W0   = (const float*)d_in[1];
  const float* b0   = (const float*)d_in[2];
  const float* W1   = (const float*)d_in[3];
  const float* b1   = (const float*)d_in[4];
  const float* Wout = (const float*)d_in[5];
  const float* bout = (const float*)d_in[6];
  float* out = (float*)d_out;

  char* ws = (char*)d_ws;
  __bf16* xb  = (__bf16*)ws; ws += (size_t)BATCH * D_IN * 2;
  __bf16* w0t = (__bf16*)ws; ws += (size_t)D_IN * HID * 2;
  __bf16* w1t = (__bf16*)ws; ws += (size_t)HID * HID * 2;
  __bf16* wot = (__bf16*)ws; ws += (size_t)HID * NSUB * 2;
  __bf16* h1  = (__bf16*)ws; ws += (size_t)BATCH * HID * 2;
  __bf16* h2  = (__bf16*)ws; ws += (size_t)BATCH * HID * 2;
  float*  sc  = (float*)ws;  ws += (size_t)BATCH * NSUB * 4;

  int n;
  n = BATCH * D_IN;
  cvt_bf16_kernel<<<(n + 255) / 256, 256, 0, stream>>>(x, xb, n);
  n = D_IN * HID;
  transpose_bf16_kernel<<<(n + 255) / 256, 256, 0, stream>>>(W0, w0t, D_IN, HID);
  n = HID * HID;
  transpose_bf16_kernel<<<(n + 255) / 256, 256, 0, stream>>>(W1, w1t, HID, HID);
  n = HID * NSUB;
  transpose_bf16_kernel<<<(n + 255) / 256, 256, 0, stream>>>(Wout, wot, HID, NSUB);

  // Layer 0: h1 = tanh(x @ W0 + b0)   [8192,512]x[512,512]
  {
    int blocks = (BATCH / 128) * (HID / 64);
    gemm_wmma_kernel<true, __bf16>
        <<<blocks, 256, 0, stream>>>(xb, w0t, b0, h1, BATCH, HID, D_IN);
  }
  // Layer 1: h2 = tanh(h1 @ W1 + b1)  [8192,512]x[512,512]
  {
    int blocks = (BATCH / 128) * (HID / 64);
    gemm_wmma_kernel<true, __bf16>
        <<<blocks, 256, 0, stream>>>(h1, w1t, b1, h2, BATCH, HID, HID);
  }
  // Layer 2: scores = h2 @ Wout + bout  [8192,512]x[512,64] (fp32 out)
  {
    int blocks = (BATCH / 128) * (NSUB / 64);
    gemm_wmma_kernel<false, float>
        <<<blocks, 256, 0, stream>>>(h2, wot, bout, sc, BATCH, NSUB, HID);
  }
  // Soft-rank + mask: one 64-thread block per batch row.
  sinkhorn_mask_kernel<<<BATCH, 64, 0, stream>>>(sc, out);
}
